// LSTMDecoder_59751585022495
// MI455X (gfx1250) — matrix operand order repaired and checked by reading, not compiled, not run on hardware
//
#include <hip/hip_runtime.h>
#include <hip/hip_bf16.h>

// LSTM greedy decoder for MI455X (gfx1250), bf16 WMMA path.
// B=32, H=E=512, V=32000, gates G=2048, T=50 (num_steps fixed per reference).

typedef __bf16 bf16_t;
typedef __attribute__((ext_vector_type(16))) __bf16 v16bf;
typedef __attribute__((ext_vector_type(8)))  __bf16 v8bf;
typedef __attribute__((ext_vector_type(8)))  float  v8f;

#define BB 32
#define HH 512
#define VV 32000
#define GG 2048
#define TT 50

// ---------------- WMMA tile helpers (wave32, contiguous k = 16*(lane/16) + s per lane) ----------------

// A-matrix 16x32 bf16 (MxK). Lane l holds row m = l&15.
// lanes 0-15: K = k0..k0+15; lanes 16-31: K = k0+16..k0+31.
__device__ __forceinline__ v16bf load_a_tile(const bf16_t* __restrict__ A, int lane,
                                             int m0, int k0, int ldk) {
  int m  = m0 + (lane & 15);
  int kk = (lane < 16) ? 0 : 16;
  const bf16_t* p = A + (size_t)m * ldk + k0 + kk;
  return *(const v16bf*)(p);
}

// B-matrix 32x16 bf16 (KxN), sourced from row-major weight W[N][K]: B[k][n] = W[n0+n][k0+k].
// Lane l holds column n = l&15; lanes 0-15: K=k0..k0+15, lanes 16-31: K=k0+16..k0+31.
__device__ __forceinline__ v16bf load_b_tile(const bf16_t* __restrict__ W, int lane,
                                             int n0, int k0, int ldk) {
  int n  = n0 + (lane & 15);
  int kk = (lane < 16) ? 0 : 16;
  return *(const v16bf*)(W + (size_t)n * ldk + k0 + kk);
}

__device__ __forceinline__ v8f wmma_bf16(v16bf a, v16bf b, v8f c) {
  return __builtin_amdgcn_wmma_f32_16x16x32_bf16(false, a, false, b, (short)0, c, false, false);
}

// ---------------- one-time preprocessing ----------------

__global__ void f2bf_kernel(const float* __restrict__ src, bf16_t* __restrict__ dst, int n) {
  int i = blockIdx.x * blockDim.x + threadIdx.x;
  if (i < n) dst[i] = (bf16_t)src[i];
}

__global__ void init_state_kernel(const float* __restrict__ hidden, bf16_t* __restrict__ h_bf,
                                  float* __restrict__ c, int* __restrict__ pred) {
  int i = blockIdx.x * blockDim.x + threadIdx.x;
  if (i < BB * HH) {
    h_bf[i] = (bf16_t)hidden[i];
    c[i] = 0.0f;
  }
  if (i < BB) pred[i] = 1;  // START_INDEX
}

// ---------------- per-step kernels ----------------

__global__ void gather_embed_kernel(const float* __restrict__ table, const int* __restrict__ pred,
                                    bf16_t* __restrict__ x_bf) {
  int b = blockIdx.x;
  const float* row = table + (size_t)pred[b] * HH;
  for (int e = threadIdx.x; e < HH; e += blockDim.x)
    x_bf[b * HH + e] = (bf16_t)row[e];
}

// gates[32,2048] = x@w_ih^T + h@w_hh^T + b_ih + b_hh   (fp32 accumulate)
// grid: 32 blocks x 256 threads (8 waves) -> 256 tiles = 2 mtiles x 128 ntiles.
__global__ void __launch_bounds__(256)
gates_gemm_kernel(const bf16_t* __restrict__ x_bf, const bf16_t* __restrict__ h_bf,
                  const bf16_t* __restrict__ wih, const bf16_t* __restrict__ whh,
                  const float* __restrict__ b_ih, const float* __restrict__ b_hh,
                  float* __restrict__ gates) {
  int wave = threadIdx.x >> 5;
  int lane = threadIdx.x & 31;
  int tile = blockIdx.x * 8 + wave;         // 0..255
  int n0 = (tile & 127) * 16;
  int m0 = (tile >> 7) * 16;

  v8f acc = {};
  for (int k0 = 0; k0 < HH; k0 += 32) {
    v16bf a = load_a_tile(x_bf, lane, m0, k0, HH);
    v16bf b = load_b_tile(wih, lane, n0, k0, HH);
    acc = wmma_bf16(a, b, acc);
  }
  for (int k0 = 0; k0 < HH; k0 += 32) {
    v16bf a = load_a_tile(h_bf, lane, m0, k0, HH);
    v16bf b = load_b_tile(whh, lane, n0, k0, HH);
    acc = wmma_bf16(a, b, acc);
  }

  int n = n0 + (lane & 15);
  float bias = b_ih[n] + b_hh[n];
  int mb = m0 + ((lane >> 4) ? 8 : 0);
  #pragma unroll
  for (int i = 0; i < 8; ++i)
    gates[(size_t)(mb + i) * GG + n] = acc[i] + bias;
}

// c_new = sigmoid(f)*c + sigmoid(i)*tanh(g); h_new = sigmoid(o)*tanh(c_new)
__global__ void lstm_update_kernel(const float* __restrict__ gates, float* __restrict__ c,
                                   bf16_t* __restrict__ h_bf) {
  int j = blockIdx.x * blockDim.x + threadIdx.x;   // 0..16383
  if (j >= BB * HH) return;
  int b = j >> 9, hh = j & (HH - 1);
  const float* g = gates + (size_t)b * GG;
  float ig = 1.0f / (1.0f + __expf(-g[hh]));
  float fg = 1.0f / (1.0f + __expf(-g[HH + hh]));
  float gg = tanhf(g[2 * HH + hh]);
  float og = 1.0f / (1.0f + __expf(-g[3 * HH + hh]));
  float cn = fg * c[j] + ig * gg;
  c[j] = cn;
  h_bf[j] = (bf16_t)(og * tanhf(cn));
}

// logits[32,32000] = h@w_out^T + b_out, written into d_out[b][t][v].
// grid: 250 blocks x 256 threads -> 2000 waves, one 16-wide N tile each, both M tiles per wave.
__global__ void __launch_bounds__(256)
logits_gemm_kernel(const bf16_t* __restrict__ h_bf, const bf16_t* __restrict__ wout,
                   const float* __restrict__ b_out, float* __restrict__ out, int t) {
  int wave = threadIdx.x >> 5;
  int lane = threadIdx.x & 31;
  int n0 = (blockIdx.x * 8 + wave) * 16;    // 0..31984

  v8f c0 = {}, c1 = {};
  for (int k0 = 0; k0 < HH; k0 += 32) {
    v16bf b  = load_b_tile(wout, lane, n0, k0, HH);  // shared by both M tiles
    v16bf a0 = load_a_tile(h_bf, lane, 0,  k0, HH);
    v16bf a1 = load_a_tile(h_bf, lane, 16, k0, HH);
    c0 = wmma_bf16(a0, b, c0);
    c1 = wmma_bf16(a1, b, c1);
  }

  int n = n0 + (lane & 15);
  float bias = b_out[n];
  int mb = ((lane >> 4) ? 8 : 0);
  #pragma unroll
  for (int i = 0; i < 8; ++i) {
    out[((size_t)(mb + i) * TT + t) * VV + n]      = c0[i] + bias;
    out[((size_t)(16 + mb + i) * TT + t) * VV + n] = c1[i] + bias;
  }
}

// First-occurrence argmax over V per batch row (matches jnp.argmax semantics).
__global__ void argmax_kernel(const float* __restrict__ out, int t, int* __restrict__ pred) {
  __shared__ float sv[256];
  __shared__ int   si[256];
  int b = blockIdx.x;
  const float* row = out + ((size_t)b * TT + t) * VV;

  float best = -__builtin_inff();
  int bi = 0;
  for (int v = threadIdx.x; v < VV; v += 256) {
    float val = row[v];
    if (val > best) { best = val; bi = v; }   // strict >: keeps earliest index per thread
  }
  sv[threadIdx.x] = best;
  si[threadIdx.x] = bi;
  __syncthreads();
  for (int s = 128; s > 0; s >>= 1) {
    if ((int)threadIdx.x < s) {
      float ov = sv[threadIdx.x + s];
      int   oi = si[threadIdx.x + s];
      if (ov > sv[threadIdx.x] || (ov == sv[threadIdx.x] && oi < si[threadIdx.x])) {
        sv[threadIdx.x] = ov;
        si[threadIdx.x] = oi;
      }
    }
    __syncthreads();
  }
  if (threadIdx.x == 0) pred[b] = si[0];
}

// ---------------- launch ----------------

extern "C" void kernel_launch(void* const* d_in, const int* in_sizes, int n_in,
                              void* d_out, int out_size, void* d_ws, size_t ws_size,
                              hipStream_t stream) {
  (void)in_sizes; (void)n_in; (void)out_size; (void)ws_size;

  const float* hidden = (const float*)d_in[0];   // [32,512]
  const float* emb    = (const float*)d_in[1];   // [32000,512]
  const float* w_ih   = (const float*)d_in[2];   // [2048,512]
  const float* w_hh   = (const float*)d_in[3];   // [2048,512]
  const float* b_ih   = (const float*)d_in[4];   // [2048]
  const float* b_hh   = (const float*)d_in[5];   // [2048]
  const float* w_out  = (const float*)d_in[6];   // [32000,512]
  const float* b_out  = (const float*)d_in[7];   // [32000]
  // d_in[8] = num_steps (fixed to 50 per reference setup)
  float* out = (float*)d_out;                    // [32,50,32000]

  // workspace carve-out (256B aligned regions)
  char* ws = (char*)d_ws;
  size_t off = 0;
  auto carve = [&](size_t bytes) -> void* {
    void* p = ws + off;
    off = (off + bytes + 255) & ~(size_t)255;
    return p;
  };
  bf16_t* wih_bf  = (bf16_t*)carve((size_t)GG * HH * sizeof(bf16_t));  // 2 MB
  bf16_t* whh_bf  = (bf16_t*)carve((size_t)GG * HH * sizeof(bf16_t));  // 2 MB
  bf16_t* wout_bf = (bf16_t*)carve((size_t)VV * HH * sizeof(bf16_t));  // 32 MB
  bf16_t* x_bf    = (bf16_t*)carve((size_t)BB * HH * sizeof(bf16_t));
  bf16_t* h_bf    = (bf16_t*)carve((size_t)BB * HH * sizeof(bf16_t));
  float*  c_f32   = (float*) carve((size_t)BB * HH * sizeof(float));
  float*  gates   = (float*) carve((size_t)BB * GG * sizeof(float));
  int*    pred    = (int*)   carve((size_t)BB * sizeof(int));

  // one-time: weights -> bf16 (after this, everything streams from the 192MB L2)
  {
    int n1 = GG * HH;
    f2bf_kernel<<<(n1 + 255) / 256, 256, 0, stream>>>(w_ih, wih_bf, n1);
    f2bf_kernel<<<(n1 + 255) / 256, 256, 0, stream>>>(w_hh, whh_bf, n1);
    int n2 = VV * HH;
    f2bf_kernel<<<(n2 + 255) / 256, 256, 0, stream>>>(w_out, wout_bf, n2);
    init_state_kernel<<<(BB * HH + 255) / 256, 256, 0, stream>>>(hidden, h_bf, c_f32, pred);
  }

  for (int t = 0; t < TT; ++t) {
    gather_embed_kernel<<<BB, 256, 0, stream>>>(emb, pred, x_bf);
    gates_gemm_kernel<<<32, 256, 0, stream>>>(x_bf, h_bf, wih_bf, whh_bf, b_ih, b_hh, gates);
    lstm_update_kernel<<<(BB * HH + 255) / 256, 256, 0, stream>>>(gates, c_f32, h_bf);
    logits_gemm_kernel<<<VV / 16 / 8, 256, 0, stream>>>(h_bf, wout_bf, b_out, out, t);
    argmax_kernel<<<BB, 256, 0, stream>>>(out, t, pred);
  }
}
